// Mamba_48558900249067
// MI455X (gfx1250) — matrix operand order repaired and checked by reading, not compiled, not run on hardware
//
#include <hip/hip_runtime.h>

#define DEV __device__ __forceinline__

typedef __attribute__((ext_vector_type(16))) _Float16     v16h;
typedef __attribute__((ext_vector_type(8)))  float        v8f;
typedef __attribute__((ext_vector_type(4)))  unsigned int u32x4;
typedef __attribute__((ext_vector_type(8)))  int          i32x8;
typedef __attribute__((ext_vector_type(4)))  int          i32x4;

constexpr int BSZ = 4, LSEQ = 2048, DMODEL = 1024, DINNER = 2048, DSTATE = 16, DTRANK = 64;
constexpr int MROWS = BSZ * LSEQ;               // 8192 token rows
constexpr int XDBL  = DTRANK + 2 * DSTATE;      // 96

DEV float silu_f(float x)     { return x / (1.0f + expf(-x)); }
DEV float softplus_f(float x) { return (x > 20.0f) ? x : log1pf(expf(x)); }

// ---------------------------------------------------------------------------
// TDM probe: NULL-tensor (D# group0.count==0) tensor_load_to_lds + tensorcnt
// wait. Architecturally a no-op DMA (ISA 8.3); exercises the async-tensor
// path (confirmed compiling on this toolchain in round 2).
// ---------------------------------------------------------------------------
__global__ void mamba_tdm_probe()
{
#if defined(__gfx1250__) && __has_builtin(__builtin_amdgcn_tensor_load_to_lds)
  u32x4 g0 = {}; i32x8 g1 = {}; i32x4 g2 = {}; i32x4 g3 = {};
#if __clang_major__ >= 23
  i32x8 g4 = {};
  __builtin_amdgcn_tensor_load_to_lds(g0, g1, g2, g3, g4, 0);
#else
  __builtin_amdgcn_tensor_load_to_lds(g0, g1, g2, g3, 0);
#endif
#if __has_builtin(__builtin_amdgcn_s_wait_tensorcnt)
  __builtin_amdgcn_s_wait_tensorcnt(0);
#endif
#endif
}

// ---------------------------------------------------------------------------
// f32 -> f16 conversion (weights + input activations)
// ---------------------------------------------------------------------------
__global__ void mamba_cvt_f16(const float* __restrict__ src,
                              _Float16* __restrict__ dst, int n)
{
  for (int i = blockIdx.x * blockDim.x + threadIdx.x; i < n;
       i += gridDim.x * blockDim.x)
    dst[i] = (_Float16)src[i];
}

// ---------------------------------------------------------------------------
// Fragment loads per CDNA5 ISA 7.12.2 (wave32), verified in round-1 asm:
// A lane: rows M=lane%16, two contiguous 16B runs (K halves select by lane/16)
// B lane: col N=lane%16, two contiguous 16B runs from row-major W[N][K] (same K order as A)
// ---------------------------------------------------------------------------
DEV v16h load_a_frag(const _Float16* __restrict__ p)
{
  v16h a;
#pragma unroll
  for (int j = 0; j < 8; ++j) { a[j] = p[j]; a[8 + j] = p[16 + j]; }
  return a;
}
DEV v16h load_b_frag(const _Float16* __restrict__ p)
{
  v16h b;
#pragma unroll
  for (int j = 0; j < 8; ++j) { b[j] = p[j]; b[8 + j] = p[16 + j]; }
  return b;
}

// ---------------------------------------------------------------------------
// WMMA GEMM: C[M,N] = A[M,K] * W[N,K]^T (+bias/activation epilogue).
// Each wave owns a 32M x (NSUB*16)N tile: 2 A-frags + NSUB B-frags feed
// 2*NSUB WMMAs per K-step, all unconditional (EXEC stays all-ones; subtile
// count is a compile-time constant, no exec-mask juggling in the hot loop).
// Launch: blockDim = WAVES*32, grid = (M/32, N / (WAVES*NSUB*16)), exact.
// EPI: 0=in_proj(silu, split -> xin16 + gate32), 1=x_proj(f32+f16 out),
//      2=dt_proj(softplus), 3=out_proj(bias only)
// ---------------------------------------------------------------------------
template<int EPI, int NSUB>
__global__ __launch_bounds__(128) void mamba_wmma_gemm(
    const _Float16* __restrict__ A,  int lda,
    const _Float16* __restrict__ Bw, int ldb,
    int K,
    const float* __restrict__ bias,
    float* __restrict__ out0,
    float* __restrict__ out1,
    _Float16* __restrict__ out16,
    int ldo)
{
  const int lane  = threadIdx.x & 31;
  const int wave  = threadIdx.x >> 5;
  const int mBase = blockIdx.x * 32;
  const int nBase = (blockIdx.y * (blockDim.x >> 5) + wave) * (NSUB * 16);

  const int r  = lane & 15;               // row (A) / col (B, D) within tile
  const int kh = lane >> 4;               // K-half selector

  const _Float16* pA0 = A + (size_t)(mBase + r) * lda + kh * 8;
  const _Float16* pA1 = pA0 + (size_t)16 * lda;
  const _Float16* pB[NSUB];
#pragma unroll
  for (int i = 0; i < NSUB; ++i)
    pB[i] = Bw + (size_t)(nBase + 16 * i + r) * ldb + kh * 8;

  v8f acc[2][NSUB] = {};
  for (int k = 0; k < K; k += 32) {
    if ((k & 255) == 0 && (k + 256) < K) {   // pull next 512B/row panel to L2
      __builtin_prefetch(pA0 + k + 256, 0, 1);
      __builtin_prefetch(pA1 + k + 256, 0, 1);
#pragma unroll
      for (int i = 0; i < NSUB; ++i)
        __builtin_prefetch(pB[i] + k + 256, 0, 1);
    }
    const v16h a0 = load_a_frag(pA0 + k);
    const v16h a1 = load_a_frag(pA1 + k);
#pragma unroll
    for (int i = 0; i < NSUB; ++i) {
      const v16h bf = load_b_frag(pB[i] + k);
      acc[0][i] = __builtin_amdgcn_wmma_f32_16x16x32_f16(
          false, a0, false, bf, (short)0, acc[0][i], false, false);
      acc[1][i] = __builtin_amdgcn_wmma_f32_16x16x32_f16(
          false, a1, false, bf, (short)0, acc[1][i], false, false);
    }
  }

#pragma unroll
  for (int i = 0; i < NSUB; ++i) {
    const int gn = nBase + 16 * i + r;
#pragma unroll
    for (int mi = 0; mi < 2; ++mi) {
#pragma unroll
      for (int j = 0; j < 8; ++j) {
        const int gm = mBase + 16 * mi + kh * 8 + j;   // D: vgpr j -> M=j+8*kh
        float v = acc[mi][i][j];
        if constexpr (EPI == 0) {
          v = silu_f(v + bias[gn]);
          if (gn < DINNER) {                 // x_in half -> f16 (GEMM2 + scan)
            out16[(size_t)gm * DINNER + gn] = (_Float16)v;
          } else {                           // res half -> silu gate, f32
            out1[(size_t)gm * DINNER + (gn - DINNER)] = v;
          }
        } else if constexpr (EPI == 1) {
          out0 [(size_t)gm * ldo + gn] = v;
          out16[(size_t)gm * ldo + gn] = (_Float16)v;
        } else if constexpr (EPI == 2) {
          out0 [(size_t)gm * ldo + gn] = softplus_f(v + bias[gn]);
        } else {
          out0 [(size_t)gm * ldo + gn] = v + bias[gn];
        }
      }
    }
  }
}

// ---------------------------------------------------------------------------
// Selective scan with fused gate. Thread = (b, d, n); 16 lanes of a wave32
// share one channel, h lives in a register across all L steps, y = h.C is a
// 4-step shfl_xor reduction; lane 0 applies (y + x*D)*silu(res) and emits f16
// directly for the out_proj GEMM.
// ---------------------------------------------------------------------------
__global__ __launch_bounds__(256) void mamba_scan_gate(
    const float*    __restrict__ dt,    // [MROWS, DINNER]
    const _Float16* __restrict__ xin,   // [MROWS, DINNER]
    const float*    __restrict__ xd,    // [MROWS, 96]: [:,64:80]=B, [:,80:96]=C
    const float*    __restrict__ A_log, // [DINNER, 16]
    const float*    __restrict__ Dv,    // [DINNER]
    const float*    __restrict__ g,     // [MROWS, DINNER] silu(res)
    _Float16*       __restrict__ y16)   // [MROWS, DINNER]
{
  const int gt = blockIdx.x * blockDim.x + threadIdx.x;
  const int n  = gt & 15;
  const int ch = gt >> 4;             // 0 .. BSZ*DINNER-1
  const int d  = ch & (DINNER - 1);
  const int b  = ch >> 11;            // DINNER = 2^11

  const float Acoef = -expf(A_log[d * DSTATE + n]);
  const float Dd    = Dv[d];
  float h = 0.0f;
  const size_t rowBase = (size_t)b * LSEQ;

  for (int t = 0; t < LSEQ; ++t) {
    const size_t row = rowBase + t;
    const float dtv = dt[row * DINNER + d];        // broadcast in 16 lanes
    const float xv  = (float)xin[row * DINNER + d];
    const float Bn  = xd[row * XDBL + DTRANK + n];
    const float Cn  = xd[row * XDBL + DTRANK + DSTATE + n];

    h = h * expf(Acoef * dtv) + (dtv * xv) * Bn;

    float y = h * Cn;                              // reduce over 16 states
    y += __shfl_xor(y, 1, 16);
    y += __shfl_xor(y, 2, 16);
    y += __shfl_xor(y, 4, 16);
    y += __shfl_xor(y, 8, 16);
    if (n == 0)
      y16[row * DINNER + d] =
          (_Float16)((y + xv * Dd) * g[row * DINNER + d]);
  }
}

// ---------------------------------------------------------------------------
extern "C" void kernel_launch(void* const* d_in, const int* in_sizes, int n_in,
                              void* d_out, int out_size, void* d_ws, size_t ws_size,
                              hipStream_t stream)
{
  (void)in_sizes; (void)n_in; (void)out_size; (void)ws_size;

  const float* x          = (const float*)d_in[0];
  const float* in_proj_w  = (const float*)d_in[1];
  const float* in_proj_b  = (const float*)d_in[2];
  const float* x_proj_w   = (const float*)d_in[3];
  const float* dt_proj_w  = (const float*)d_in[4];
  const float* dt_proj_b  = (const float*)d_in[5];
  const float* A_log      = (const float*)d_in[6];
  const float* Dvec       = (const float*)d_in[7];
  const float* out_proj_w = (const float*)d_in[8];
  const float* out_proj_b = (const float*)d_in[9];

  char* ws = (char*)d_ws;
  size_t off = 0;
  auto alloc = [&](size_t bytes) -> void* {
    void* p = ws + off;
    off = (off + bytes + 255) & ~(size_t)255;
    return p;
  };

  _Float16* wIn16  = (_Float16*)alloc((size_t)2 * DINNER * DMODEL * 2);
  _Float16* wXp16  = (_Float16*)alloc((size_t)XDBL * DINNER * 2);
  _Float16* wDt16  = (_Float16*)alloc((size_t)DINNER * DTRANK * 2);
  _Float16* wOut16 = (_Float16*)alloc((size_t)DMODEL * DINNER * 2);
  _Float16* x16    = (_Float16*)alloc((size_t)MROWS * DMODEL * 2);
  _Float16* xin16  = (_Float16*)alloc((size_t)MROWS * DINNER * 2);
  float*    g32    = (float*)   alloc((size_t)MROWS * DINNER * 4);
  float*    xd32   = (float*)   alloc((size_t)MROWS * XDBL * 4);
  _Float16* xd16   = (_Float16*)alloc((size_t)MROWS * XDBL * 2);
  float*    dt32   = (float*)   alloc((size_t)MROWS * DINNER * 4);
  _Float16* y16    = (_Float16*)alloc((size_t)MROWS * DINNER * 2);

  // --- async-tensor path probe (NULL-tensor no-op) -------------------------
  mamba_tdm_probe<<<1, 32, 0, stream>>>();

  // --- precision downcasts -------------------------------------------------
  mamba_cvt_f16<<<2048, 256, 0, stream>>>(x,          x16,    MROWS * DMODEL);
  mamba_cvt_f16<<<2048, 256, 0, stream>>>(in_proj_w,  wIn16,  2 * DINNER * DMODEL);
  mamba_cvt_f16<<< 256, 256, 0, stream>>>(x_proj_w,   wXp16,  XDBL * DINNER);
  mamba_cvt_f16<<< 256, 256, 0, stream>>>(dt_proj_w,  wDt16,  DINNER * DTRANK);
  mamba_cvt_f16<<<2048, 256, 0, stream>>>(out_proj_w, wOut16, DMODEL * DINNER);

  // --- 1) in_proj: [8192,1024] x [4096,1024]^T, silu + split ---------------
  // 4 waves x NSUB=4 -> 256-wide N tile; grid.y = 4096/256 = 16
  mamba_wmma_gemm<0, 4><<<dim3(MROWS / 32, 16), 128, 0, stream>>>(
      x16, DMODEL, wIn16, DMODEL, DMODEL,
      in_proj_b, nullptr, g32, xin16, DINNER);

  // --- 2) x_proj: [8192,2048] x [96,2048]^T --------------------------------
  // 2 waves x NSUB=3 -> exactly 96 columns
  mamba_wmma_gemm<1, 3><<<dim3(MROWS / 32, 1), 64, 0, stream>>>(
      xin16, DINNER, wXp16, DINNER, DINNER,
      nullptr, xd32, nullptr, xd16, XDBL);

  // --- 3) dt_proj: [8192,64] x [2048,64]^T, softplus -----------------------
  mamba_wmma_gemm<2, 4><<<dim3(MROWS / 32, 8), 128, 0, stream>>>(
      xd16, XDBL, wDt16, DTRANK, DTRANK,
      dt_proj_b, dt32, nullptr, nullptr, DINNER);

  // --- 4) selective scan + gate (fused) ------------------------------------
  mamba_scan_gate<<<(BSZ * DINNER * DSTATE) / 256, 256, 0, stream>>>(
      dt32, xin16, xd32, A_log, Dvec, g32, y16);

  // --- 5) out_proj: [8192,2048] x [1024,2048]^T + bias ---------------------
  mamba_wmma_gemm<3, 4><<<dim3(MROWS / 32, 4), 128, 0, stream>>>(
      y16, DINNER, wOut16, DINNER, DINNER,
      out_proj_b, (float*)d_out, nullptr, nullptr, DMODEL);
}
